// SelfAttention_32427003085585
// MI455X (gfx1250) — compile-verified, matrix operand order checked by reading, not run on hardware
//
#include <hip/hip_runtime.h>

typedef __attribute__((ext_vector_type(16))) _Float16 v16h;
typedef __attribute__((ext_vector_type(8)))  _Float16 v8h;
typedef __attribute__((ext_vector_type(4)))  _Float16 v4h;
typedef __attribute__((ext_vector_type(8)))  float    v8f;
typedef __attribute__((ext_vector_type(4)))  unsigned int u32x4;
typedef __attribute__((ext_vector_type(8)))  unsigned int u32x8;

#define WMMA_F16(a, b, c) \
  __builtin_amdgcn_wmma_f32_16x16x32_f16(false, (a), false, (b), (short)0, (c), false, false)

constexpr int Bb = 2, Ss = 2048, Dd = 2048, Hh = 16, HDd = 128;
constexpr float SCALE = 0.08838834764831843f;  // 128^-0.5

// 16x32 f16 A-fragment (or 32x16 B-fragment from a transposed LDS tile), per
// CDNA5 ISA layout: lane%16 = M/N row; lane-half selects K base 0 vs 8;
// halves 0..7 = K base+0..7, halves 8..15 = K base+16..23 (two ds_load_b128).
__device__ __forceinline__ v16h frag_ld(const _Float16* p0, int row, int stride, int kbase) {
  const int lane = threadIdx.x & 31;
  const _Float16* p = p0 + row * stride + kbase + ((lane & 16) ? 8 : 0);
  union { v16h v; v8h h[2]; } u;
  u.h[0] = *(const v8h*)(p);
  u.h[1] = *(const v8h*)(p + 16);
  return u.v;
}

// C = A(MxK) @ W(KxN) + bias.  Block tile 128x128, K-step 32, 8 waves,
// wave tile 64x32 (4x2 WMMA frags), register double-buffered staging.
// AF16: A is f16 workspace [M,K]; else f32 global [M,K].
// EPI 0: store f16 to [B,H,S,HD] (QKV split-heads). EPI 1: store f32 [M,N].
template <bool AF16, int EPI>
__global__ __launch_bounds__(256) void gemm128(
    const void* __restrict__ Ap, const float* __restrict__ W,
    const float* __restrict__ bias, void* __restrict__ Cp,
    int M, int N, int K) {
  __shared__ _Float16 As[128 * 40];   // As[m][k], stride 40
  __shared__ _Float16 Bt[128 * 40];   // Bt[n][k], stride 40 (W transposed)
  const int t = threadIdx.x;
  const int lane = t & 31, wave = t >> 5;
  const int wm = wave >> 2, wn = wave & 3;
  const int row0 = blockIdx.y * 128, col0 = blockIdx.x * 128;

  v8f acc[4][2];
#pragma unroll
  for (int mt = 0; mt < 4; ++mt)
#pragma unroll
    for (int nt = 0; nt < 2; ++nt)
#pragma unroll
      for (int e = 0; e < 8; ++e) acc[mt][nt][e] = 0.0f;

  float4 aF[4];
  v8h    aH[2];
  float4 wF[4];

  auto fetch = [&](int kk) {
    if (AF16) {
      const _Float16* A16 = (const _Float16*)Ap;
#pragma unroll
      for (int i = 0; i < 2; ++i) {
        const int r = (t >> 2) + 64 * i;
        const int cg = (t & 3) * 8;
        aH[i] = *(const v8h*)&A16[(size_t)(row0 + r) * K + kk + cg];
      }
    } else {
      const float* A32 = (const float*)Ap;
#pragma unroll
      for (int i = 0; i < 4; ++i) {
        const int r = (t >> 3) + 32 * i;
        const int cg = (t & 7) * 4;
        aF[i] = *(const float4*)&A32[(size_t)(row0 + r) * K + kk + cg];
      }
    }
#pragma unroll
    for (int i = 0; i < 4; ++i) {
      const int kr = (t >> 5) + 8 * i;
      const int cg = (t & 31) * 4;
      wF[i] = *(const float4*)&W[(size_t)(kk + kr) * N + col0 + cg];
    }
  };

  fetch(0);
  for (int kk = 0; kk < K; kk += 32) {
    // stage prefetched tile into LDS (f16)
    if (AF16) {
#pragma unroll
      for (int i = 0; i < 2; ++i) {
        const int r = (t >> 2) + 64 * i;
        const int cg = (t & 3) * 8;
        *(v8h*)&As[r * 40 + cg] = aH[i];
      }
    } else {
#pragma unroll
      for (int i = 0; i < 4; ++i) {
        const int r = (t >> 3) + 32 * i;
        const int cg = (t & 7) * 4;
        v4h h;
        h.x = (_Float16)aF[i].x; h.y = (_Float16)aF[i].y;
        h.z = (_Float16)aF[i].z; h.w = (_Float16)aF[i].w;
        *(v4h*)&As[r * 40 + cg] = h;
      }
    }
#pragma unroll
    for (int i = 0; i < 4; ++i) {
      const int kr = (t >> 5) + 8 * i;
      const int cg = (t & 31) * 4;
      Bt[(cg + 0) * 40 + kr] = (_Float16)wF[i].x;
      Bt[(cg + 1) * 40 + kr] = (_Float16)wF[i].y;
      Bt[(cg + 2) * 40 + kr] = (_Float16)wF[i].z;
      Bt[(cg + 3) * 40 + kr] = (_Float16)wF[i].w;
    }
    __syncthreads();
    if (kk + 32 < K) fetch(kk + 32);  // overlap global latency with WMMA burst

    v16h a[4], b[2];
#pragma unroll
    for (int mt = 0; mt < 4; ++mt) a[mt] = frag_ld(As, wm * 64 + mt * 16 + (lane & 15), 40, 0);
#pragma unroll
    for (int nt = 0; nt < 2; ++nt) b[nt] = frag_ld(Bt, wn * 32 + nt * 16 + (lane & 15), 40, 0);
#pragma unroll
    for (int mt = 0; mt < 4; ++mt)
#pragma unroll
      for (int nt = 0; nt < 2; ++nt) acc[mt][nt] = WMMA_F16(a[mt], b[nt], acc[mt][nt]);
    __syncthreads();
  }

#pragma unroll
  for (int mt = 0; mt < 4; ++mt) {
#pragma unroll
    for (int nt = 0; nt < 2; ++nt) {
      const int n = col0 + wn * 32 + nt * 16 + (lane & 15);
      const float bv = bias[n];
#pragma unroll
      for (int r = 0; r < 8; ++r) {
        const int m = row0 + wm * 64 + mt * 16 + r + ((lane >> 4) << 3);
        const float v = acc[mt][nt][r] + bv;
        if (EPI == 0) {  // -> f16 [B,H,S,HD]
          _Float16* o = (_Float16*)Cp;
          const int bi = m >> 11, s = m & 2047;       // S = 2048
          const int h = n >> 7, hd = n & 127;         // HD = 128
          o[(((size_t)bi * Hh + h) * Ss + s) * HDd + hd] = (_Float16)v;
        } else {         // -> f32 [M,N]
          ((float*)Cp)[(size_t)m * N + n] = v;
        }
      }
    }
  }
}

// Partial interleaved rotary on q,k first 32 dims. One thread per pair.
__global__ __launch_bounds__(256) void rope_kernel(
    _Float16* __restrict__ q, _Float16* __restrict__ k,
    const float* __restrict__ cs, const float* __restrict__ sn) {
  const int idx = blockIdx.x * 256 + threadIdx.x;  // B*H*S*16 pairs
  const int i = idx & 15;
  const int s = (idx >> 4) & 2047;
  const int bh = idx >> 15;
  const size_t off = (size_t)bh * Ss * HDd + (size_t)s * HDd + 2 * i;
  const float c0 = cs[s * 32 + 2 * i], c1 = cs[s * 32 + 2 * i + 1];
  const float s0 = sn[s * 32 + 2 * i], s1 = sn[s * 32 + 2 * i + 1];
  float a = (float)q[off], b = (float)q[off + 1];
  q[off]     = (_Float16)(a * c0 - b * s0);
  q[off + 1] = (_Float16)(b * c1 + a * s1);
  a = (float)k[off]; b = (float)k[off + 1];
  k[off]     = (_Float16)(a * c0 - b * s0);
  k[off + 1] = (_Float16)(b * c1 + a * s1);
}

// Flash attention, non-causal. Grid (S/64, H, B); 4 waves x 16 q-rows.
// K tiles: TDM tensor_load_to_lds, double-buffered, padded rows (stride 136),
// gated by TENSORcnt + barrier.  V tiles: register prefetch + LDS transpose.
__global__ __launch_bounds__(128) void attn_kernel(
    const _Float16* __restrict__ Q, const _Float16* __restrict__ Kw,
    const _Float16* __restrict__ Vw, _Float16* __restrict__ Ctx) {
  __shared__ _Float16 Qs[64 * 136];      // [qrow][d]
  __shared__ _Float16 Ks[2][64 * 136];   // [key][d] == Bt for Q.K^T (TDM dest)
  __shared__ _Float16 Vt[128 * 72];      // [d][key] == Bt for P.V
  __shared__ _Float16 Ps[4 * 16 * 72];   // per-wave P scratch [qrow][key]
  const int t = threadIdx.x, lane = t & 31, w = t >> 5;
  const int q0 = blockIdx.x * 64;
  const size_t bh = ((size_t)blockIdx.z * Hh + blockIdx.y) * Ss * HDd;

  // TDM: DMA one 64x128 f16 K-tile into Ks[buf] with 8-half row padding.
  auto tdm_load_k = [&](int j, int buf) {
    const unsigned long long ga =
        (unsigned long long)(uintptr_t)(Kw + bh + (size_t)(j * 64) * HDd);
    const unsigned int la = (unsigned int)(uintptr_t)&Ks[buf][0];
    u32x4 g0;
    g0[0] = 1u;                                   // count=1 (valid descriptor)
    g0[1] = la;                                   // lds_addr
    g0[2] = (unsigned int)ga;                     // global_addr[31:0]
    g0[3] = (unsigned int)((ga >> 32) & 0x1FFFFFFu) | (2u << 30);  // [56:32] | type=2
    u32x8 g1;
    g1[0] = (1u << 16)        // data_size = 2 bytes
          | (1u << 20)        // pad_enable
          | (5u << 22)        // pad_interval: 64 DWORDs (one 128-half row)
          | (3u << 25);       // pad_amount: 4 DWORDs (8 halves) -> stride 136
    g1[1] = 128u << 16;       // tensor_dim0 = 128 (bits 79:48, low half)
    g1[2] = 64u << 16;        // tensor_dim0 hi=0 | tensor_dim1 = 64 (low half)
    g1[3] = 128u << 16;       // tensor_dim1 hi=0 | tile_dim0 = 128
    g1[4] = 64u;              // tile_dim1 = 64, tile_dim2 = 0
    g1[5] = 128u;             // tensor_dim0_stride = 128 (48-bit, low dword)
    g1[6] = 0u;               // stride hi | tensor_dim1_stride lo
    g1[7] = 0u;
    asm volatile("tensor_load_to_lds %0, %1" :: "s"(g0), "s"(g1) : "memory");
  };

  // stage Q once
#pragma unroll
  for (int i = 0; i < 8; ++i) {
    const int r = (t >> 4) + 8 * i;
    const int cg = (t & 15) * 8;
    *(v8h*)&Qs[r * 136 + cg] = *(const v8h*)&Q[bh + (size_t)(q0 + r) * HDd + cg];
  }

  v8h vreg[8], vnext[8];
  auto fetch_v = [&](int j, v8h* dst) {
#pragma unroll
    for (int i = 0; i < 8; ++i) {
      const int r = (t >> 4) + 8 * i;
      const int cg = (t & 15) * 8;
      dst[i] = *(const v8h*)&Vw[bh + (size_t)(j * 64 + r) * HDd + cg];
    }
  };

  if (w == 0) tdm_load_k(0, 0);   // kick off first K tile on the TDM
  fetch_v(0, vreg);

  __syncthreads();
  v16h aQ[4];
#pragma unroll
  for (int c = 0; c < 4; ++c) aQ[c] = frag_ld(Qs, w * 16 + (lane & 15), 136, c * 32);

  float mrow[8], lrow[8];
  v8f ctx[8];
#pragma unroll
  for (int r = 0; r < 8; ++r) { mrow[r] = -3.0e38f; lrow[r] = 0.0f; }
#pragma unroll
  for (int d = 0; d < 8; ++d)
#pragma unroll
    for (int e = 0; e < 8; ++e) ctx[d][e] = 0.0f;

  for (int j = 0; j < Ss / 64; ++j) {
    const int buf = j & 1;
    if (w == 0) __builtin_amdgcn_s_wait_tensorcnt(0);  // K[buf] landed in LDS
    __syncthreads();   // broadcast readiness; prior-iter Ks/Vt reads complete

    // V tile (prefetched in regs) -> transposed LDS
#pragma unroll
    for (int i = 0; i < 8; ++i) {
      const int r = (t >> 4) + 8 * i;
      const int cg = (t & 15) * 8;
#pragma unroll
      for (int e = 0; e < 8; ++e) Vt[(cg + e) * 72 + r] = vreg[i][e];
    }
    if (j + 1 < Ss / 64) {
      if (w == 0) tdm_load_k(j + 1, buf ^ 1);  // TDM overlaps with compute
      fetch_v(j + 1, vnext);                   // VMEM overlaps with compute
    }
    __syncthreads();

    const _Float16* Kbuf = &Ks[buf][0];
    v8f sc[4];
#pragma unroll
    for (int nt = 0; nt < 4; ++nt) {
      v8f a;
#pragma unroll
      for (int e = 0; e < 8; ++e) a[e] = 0.0f;
#pragma unroll
      for (int c = 0; c < 4; ++c) {
        const v16h bK = frag_ld(Kbuf, nt * 16 + (lane & 15), 136, c * 32);
        a = WMMA_F16(aQ[c], bK, a);
      }
#pragma unroll
      for (int e = 0; e < 8; ++e) sc[nt][e] = a[e] * SCALE;
    }

    float mnew[8], corr[8], rs[8];
#pragma unroll
    for (int r = 0; r < 8; ++r) {
      float tm = fmaxf(fmaxf(sc[0][r], sc[1][r]), fmaxf(sc[2][r], sc[3][r]));
      tm = fmaxf(tm, __shfl_xor(tm, 1, 32));
      tm = fmaxf(tm, __shfl_xor(tm, 2, 32));
      tm = fmaxf(tm, __shfl_xor(tm, 4, 32));
      tm = fmaxf(tm, __shfl_xor(tm, 8, 32));  // masks<16 stay in the row-sharing half
      mnew[r] = fmaxf(mrow[r], tm);
      corr[r] = __expf(mrow[r] - mnew[r]);
      mrow[r] = mnew[r];
      rs[r] = 0.0f;
    }
#pragma unroll
    for (int d = 0; d < 8; ++d)
#pragma unroll
      for (int r = 0; r < 8; ++r) ctx[d][r] *= corr[r];

    _Float16* Pw = &Ps[w * 16 * 72];
#pragma unroll
    for (int nt = 0; nt < 4; ++nt)
#pragma unroll
      for (int r = 0; r < 8; ++r) {
        const float p = __expf(sc[nt][r] - mnew[r]);
        rs[r] += p;
        Pw[(r + ((lane >> 4) << 3)) * 72 + nt * 16 + (lane & 15)] = (_Float16)p;
      }
#pragma unroll
    for (int r = 0; r < 8; ++r) {
      float s2 = rs[r];
      s2 += __shfl_xor(s2, 1, 32);
      s2 += __shfl_xor(s2, 2, 32);
      s2 += __shfl_xor(s2, 4, 32);
      s2 += __shfl_xor(s2, 8, 32);
      lrow[r] = lrow[r] * corr[r] + s2;
    }
    asm volatile("s_wait_dscnt 0" ::: "memory");  // P stores visible to own wave
#pragma unroll
    for (int c2 = 0; c2 < 2; ++c2) {
      const v16h aP = frag_ld(Pw, (lane & 15), 72, c2 * 32);
#pragma unroll
      for (int d = 0; d < 8; ++d) {
        const v16h bV = frag_ld(Vt, d * 16 + (lane & 15), 72, c2 * 32);
        ctx[d] = WMMA_F16(aP, bV, ctx[d]);
      }
    }
#pragma unroll
    for (int i = 0; i < 8; ++i) vreg[i] = vnext[i];
  }

#pragma unroll
  for (int r = 0; r < 8; ++r) {
    const float inv = 1.0f / lrow[r];
    const int s = q0 + w * 16 + r + ((lane >> 4) << 3);
    const size_t base = ((size_t)blockIdx.z * Ss + s) * Dd + (size_t)blockIdx.y * HDd;
#pragma unroll
    for (int d = 0; d < 8; ++d)
      Ctx[base + d * 16 + (lane & 15)] = (_Float16)(ctx[d][r] * inv);
  }
}

extern "C" void kernel_launch(void* const* d_in, const int* in_sizes, int n_in,
                              void* d_out, int out_size, void* d_ws, size_t ws_size,
                              hipStream_t stream) {
  const float* hs = (const float*)d_in[0];
  const float* cs = (const float*)d_in[1];
  const float* sn = (const float*)d_in[2];
  const float* Wq = (const float*)d_in[3];
  const float* bq = (const float*)d_in[4];
  const float* Wk = (const float*)d_in[5];
  const float* bk = (const float*)d_in[6];
  const float* Wv = (const float*)d_in[7];
  const float* bv = (const float*)d_in[8];
  const float* Wo = (const float*)d_in[9];
  const float* bo = (const float*)d_in[10];

  const size_t MD = (size_t)Bb * Ss * Dd;  // 8M elements
  _Float16* q16 = (_Float16*)d_ws;
  _Float16* k16 = q16 + MD;
  _Float16* v16 = k16 + MD;
  _Float16* c16 = v16 + MD;

  const dim3 gp(Dd / 128, (Bb * Ss) / 128, 1);
  gemm128<false, 0><<<gp, 256, 0, stream>>>(hs, Wq, bq, q16, Bb * Ss, Dd, Dd);
  gemm128<false, 0><<<gp, 256, 0, stream>>>(hs, Wk, bk, k16, Bb * Ss, Dd, Dd);
  gemm128<false, 0><<<gp, 256, 0, stream>>>(hs, Wv, bv, v16, Bb * Ss, Dd, Dd);
  rope_kernel<<<(Bb * Hh * Ss * 16) / 256, 256, 0, stream>>>(q16, k16, cs, sn);
  attn_kernel<<<dim3(Ss / 64, Hh, Bb), 128, 0, stream>>>(q16, k16, v16, c16);
  gemm128<true, 1><<<gp, 256, 0, stream>>>(c16, Wo, bo, d_out, Bb * Ss, Dd, Dd);
}
